// KimiDeltaAttention_74371653698287
// MI455X (gfx1250) — matrix operand order repaired and checked by reading, not compiled, not run on hardware
//
#include <hip/hip_runtime.h>
#include <math.h>

#define B_    2
#define T_    2048
#define HID_  2048
#define H_    16
#define D_    128
#define P_    2048
#define EPS_  1e-6f

typedef __attribute__((ext_vector_type(8)))  __bf16 v8bf;
typedef __attribute__((ext_vector_type(16))) __bf16 v16bf;
typedef __attribute__((ext_vector_type(8)))  float  v8f;
typedef __attribute__((ext_vector_type(4)))  float  v4f;

__device__ __forceinline__ __bf16 f2bf(float f) {
    union { float f; unsigned u; } x; x.f = f;
    unsigned r = (x.u + 0x7FFFu + ((x.u >> 16) & 1u)) >> 16;   // RNE
    union { unsigned short s; __bf16 b; } y; y.s = (unsigned short)r;
    return y.b;
}

__device__ __forceinline__ float sigmoidf_(float x) {
    return 1.0f / (1.0f + __expf(-x));
}

// ---------------------------------------------------------------------------
// fp32 -> bf16 pack (8 elements / thread). n must be a multiple of 8.
// ---------------------------------------------------------------------------
__global__ void f32_to_bf16_kernel(const float* __restrict__ x,
                                   __bf16* __restrict__ y, long n)
{
    long i = ((long)blockIdx.x * blockDim.x + threadIdx.x) * 8;
    if (i >= n) return;
    v4f x0 = *(const v4f*)(x + i);
    v4f x1 = *(const v4f*)(x + i + 4);
    v8bf r;
    #pragma unroll
    for (int j = 0; j < 4; ++j) { r[j] = f2bf(x0[j]); r[4 + j] = f2bf(x1[j]); }
    *(v8bf*)(y + i) = r;
}

// ---------------------------------------------------------------------------
// GEMM: C[M,N] = A[M,Kd] * W[N,Kd]^T   (bf16 in, fp32 accum/out via WMMA)
// Block = 256 threads = 8 waves. Each wave: 16 rows x NT*16 cols (NT tiles).
// The (NT*16)x32 W tile is shared by all 8 waves -> staged once in LDS.
// NT is compile time: no branches in the k-loop, accumulators stay pinned.
// grid.x = M/128, grid.y = N/(NT*16). Kd % 32 == 0.
// ---------------------------------------------------------------------------
#define BT_STRIDE 40   // padded row stride (bf16) to break bank conflicts
template <int NT>
__global__ void gemm_bf16_wmma(const __bf16* __restrict__ A,
                               const __bf16* __restrict__ W,
                               float* __restrict__ Cout,
                               int M, int N, int Kd)
{
    __shared__ __bf16 Bt[NT * 16][BT_STRIDE];

    const int tid    = threadIdx.x;
    const int wave   = tid >> 5;
    const int lane   = tid & 31;
    const int laneLo = lane & 15;
    const int laneHi = lane >> 4;              // 0 or 1
    const int m0     = blockIdx.x * 128 + wave * 16;
    const int n0     = blockIdx.y * (NT * 16);

    // cooperative B-tile loader: 4 threads/row, 8 bf16 (16B) each
    const int  ldRow = tid >> 2;               // 0..63
    const int  ldCol = (tid & 3) * 8;          // 0,8,16,24
    const bool ldOk  = ldRow < NT * 16;
    const __bf16* wsrc = W + (size_t)(n0 + ldRow) * Kd + ldCol;

    v8f acc[NT];
    #pragma unroll
    for (int j = 0; j < NT; ++j) acc[j] = (v8f){0,0,0,0,0,0,0,0};

    const __bf16* arow = A + (size_t)(m0 + laneLo) * Kd;

    for (int k0 = 0; k0 < Kd; k0 += 32) {
        // ---- stage W[n0 .. n0+NT*16-1][k0..k0+31] into LDS ----
        if (ldOk)
            *(v8bf*)&Bt[ldRow][ldCol] = *(const v8bf*)(wsrc + k0);
        __syncthreads();

        // ---- A fragment (16x32 bf16, ISA lane layout) from global ----
        const int kb = k0 + laneHi * 8;        // lo lanes: K 0..7 ; hi: 8..15
        v8bf a0 = *(const v8bf*)(arow + kb);
        v8bf a1 = *(const v8bf*)(arow + kb + 16);
        v16bf afrag = __builtin_shufflevector(a0, a1,
            0, 1, 2, 3, 4, 5, 6, 7, 8, 9, 10, 11, 12, 13, 14, 15);

        // ---- NT B fragments (32x16 bf16) from LDS, branch-free ----
        #pragma unroll
        for (int j = 0; j < NT; ++j) {
            const __bf16* brow = &Bt[j * 16 + laneLo][laneHi * 8];
            v8bf b0 = *(const v8bf*)(brow);
            v8bf b1 = *(const v8bf*)(brow + 16);
            v16bf bfrag = __builtin_shufflevector(b0, b1,
                0, 1, 2, 3, 4, 5, 6, 7, 8, 9, 10, 11, 12, 13, 14, 15);
            acc[j] = __builtin_amdgcn_wmma_f32_16x16x32_bf16(
                false, afrag, false, bfrag, (short)0, acc[j], false, false);
        }
        __syncthreads();
    }

    // ---- store: VGPR r -> row m0 + laneHi*8 + r, col n0 + j*16 + laneLo ----
    #pragma unroll
    for (int j = 0; j < NT; ++j) {
        const int n    = n0 + j * 16 + laneLo;
        const int mrow = m0 + laneHi * 8;
        #pragma unroll
        for (int r = 0; r < 8; ++r)
            Cout[(size_t)(mrow + r) * N + n] = acc[j][r];
    }
}

// ---------------------------------------------------------------------------
// Depthwise causal conv (K=4) + SiLU over [B,T,P]; w: [P,4]
// ---------------------------------------------------------------------------
__global__ void conv_silu_kernel(const float* __restrict__ x,
                                 const float* __restrict__ w,
                                 float* __restrict__ y)
{
    const size_t total = (size_t)B_ * T_ * P_;
    size_t idx = (size_t)blockIdx.x * blockDim.x + threadIdx.x;
    if (idx >= total) return;
    const int c = (int)(idx % P_);
    const int t = (int)((idx / P_) % T_);
    float acc = 0.0f;
    #pragma unroll
    for (int j = 0; j < 4; ++j) {
        const int ts = t - 3 + j;
        if (ts >= 0)
            acc += x[(long)idx + (long)(ts - t) * P_] * w[c * 4 + j];
    }
    y[idx] = acc * sigmoidf_(acc);
}

// ---------------------------------------------------------------------------
// g = -exp(A_log[h]) * softplus(glin + dt_bias)   (in-place, [B,T,P])
// ---------------------------------------------------------------------------
__global__ void kda_gate_kernel(float* __restrict__ g,
                                const float* __restrict__ dt_bias,
                                const float* __restrict__ A_log)
{
    const size_t total = (size_t)B_ * T_ * P_;
    size_t idx = (size_t)blockIdx.x * blockDim.x + threadIdx.x;
    if (idx >= total) return;
    const int p = (int)(idx % P_);
    const int h = p >> 7;
    float x = g[idx] + dt_bias[p];
    float sp = (x > 20.0f) ? x : log1pf(__expf(x));
    g[idx] = -__expf(A_log[h]) * sp;
}

__global__ void sigmoid_kernel(float* __restrict__ x, int n)
{
    int idx = blockIdx.x * blockDim.x + threadIdx.x;
    if (idx < n) x[idx] = sigmoidf_(x[idx]);
}

// ---------------------------------------------------------------------------
// L2 norm over last dim (128) in-place, with scale. grid = B*T*H, block = 128.
// ---------------------------------------------------------------------------
__global__ void l2norm_kernel(float* __restrict__ buf, float scale)
{
    __shared__ float red[128];
    const size_t base = (size_t)blockIdx.x * 128;
    const int tid = threadIdx.x;
    float v = buf[base + tid];
    red[tid] = v * v;
    __syncthreads();
    for (int s = 64; s > 0; s >>= 1) {
        if (tid < s) red[tid] += red[tid + s];
        __syncthreads();
    }
    buf[base + tid] = v * rsqrtf(red[0] + EPS_) * scale;
}

// ---------------------------------------------------------------------------
// Gated delta-rule recurrence. grid = B*H, block = 256.
// State S[k][v] (128x128) in registers: thread (col=tid&127, half=tid>>7)
// owns S[half*64 .. half*64+63][col].
// ---------------------------------------------------------------------------
__global__ void recurrence_kernel(const float* __restrict__ q,
                                  const float* __restrict__ k,
                                  const float* __restrict__ v,
                                  const float* __restrict__ g,
                                  const float* __restrict__ beta,
                                  float* __restrict__ o)
{
    const int bh = blockIdx.x;
    const int b  = bh / H_;
    const int h  = bh % H_;
    const int tid   = threadIdx.x;
    const int col   = tid & 127;
    const int half  = tid >> 7;
    const int kbase = half * 64;

    __shared__ float sk[128], sq[128], sv[128], sg[128];
    __shared__ float pmem[2][128], pout[2][128], sdelta[128];
    __shared__ float sbeta;

    float S[64];
    #pragma unroll
    for (int i = 0; i < 64; ++i) S[i] = 0.0f;

    const size_t base = ((size_t)b * T_) * P_ + (size_t)h * D_;

    for (int t = 0; t < T_; ++t) {
        const size_t off = base + (size_t)t * P_;
        if (tid < 128) {
            sk[tid] = k[off + tid];
            sq[tid] = q[off + tid];
            sv[tid] = v[off + tid];
            sg[tid] = __expf(g[off + tid]);
        }
        if (tid == 0) sbeta = beta[((size_t)b * T_ + t) * H_ + h];
        __syncthreads();

        float pm = 0.0f;
        #pragma unroll
        for (int i = 0; i < 64; ++i) {
            S[i] *= sg[kbase + i];
            pm += sk[kbase + i] * S[i];
        }
        pmem[half][col] = pm;
        __syncthreads();

        if (half == 0) {
            const float mem = pmem[0][col] + pmem[1][col];
            sdelta[col] = (sv[col] - mem) * sbeta;
        }
        __syncthreads();

        const float d = sdelta[col];
        float po = 0.0f;
        #pragma unroll
        for (int i = 0; i < 64; ++i) {
            S[i] += sk[kbase + i] * d;
            po += sq[kbase + i] * S[i];
        }
        pout[half][col] = po;
        __syncthreads();

        if (half == 0) o[off + col] = pout[0][col] + pout[1][col];
        __syncthreads();
    }
}

// ---------------------------------------------------------------------------
// RMS norm over D (in-place) * o_norm_w * sigmoid(gout). grid=B*T*H, block=128
// ---------------------------------------------------------------------------
__global__ void normgate_kernel(float* __restrict__ o,
                                const float* __restrict__ gout,
                                const float* __restrict__ w)
{
    __shared__ float red[128];
    const size_t base = (size_t)blockIdx.x * 128;
    const int tid = threadIdx.x;
    float v = o[base + tid];
    red[tid] = v * v;
    __syncthreads();
    for (int s = 64; s > 0; s >>= 1) {
        if (tid < s) red[tid] += red[tid + s];
        __syncthreads();
    }
    const float rn = rsqrtf(red[0] * (1.0f / 128.0f) + EPS_);
    o[base + tid] = v * rn * w[tid] * sigmoidf_(gout[base + tid]);
}

// ---------------------------------------------------------------------------
extern "C" void kernel_launch(void* const* d_in, const int* in_sizes, int n_in,
                              void* d_out, int out_size, void* d_ws, size_t ws_size,
                              hipStream_t stream)
{
    const float* hs      = (const float*)d_in[0];
    const float* Wq      = (const float*)d_in[1];
    const float* Wk      = (const float*)d_in[2];
    const float* Wv      = (const float*)d_in[3];
    const float* wq_conv = (const float*)d_in[4];
    const float* wk_conv = (const float*)d_in[5];
    const float* wv_conv = (const float*)d_in[6];
    const float* A_log   = (const float*)d_in[7];
    const float* Wfa     = (const float*)d_in[8];
    const float* Wfb     = (const float*)d_in[9];
    const float* dt_bias = (const float*)d_in[10];
    const float* Wb      = (const float*)d_in[11];
    const float* Wga     = (const float*)d_in[12];
    const float* Wgb     = (const float*)d_in[13];
    const float* o_norm_w= (const float*)d_in[14];
    const float* Wo      = (const float*)d_in[15];
    float* out = (float*)d_out;
    char*  base = (char*)d_ws;

    const int    M    = B_ * T_;                    // 4096
    const long   SZ   = (long)M * P_;               // 8M elements
    const long   SZW  = (long)P_ * HID_;            // 4M (big weights)
    const long   SZS  = (long)P_ * D_;              // 256K (small weights)
    const long   SZF  = (long)M * D_;               // 512K (fa/ga)
    const size_t FB   = (size_t)SZ * 4;             // 32MB

    // fp32 workspace
    float* bufA = (float*)(base);                   // qlin -> gout
    float* bufB = (float*)(base + 1 * FB);          // klin -> o
    float* bufC = (float*)(base + 2 * FB);          // vlin -> g
    float* qc   = (float*)(base + 3 * FB);
    float* kc   = (float*)(base + 4 * FB);
    float* vc   = (float*)(base + 5 * FB);
    float* fbuf = (float*)(base + 6 * FB);                      // M*D
    float* bbuf = (float*)(base + 6 * FB + (size_t)SZF * 4);    // M*H
    // bf16 workspace
    char*  bp     = base + 6 * FB + (size_t)SZF * 4 + (size_t)M * H_ * 4;
    __bf16* hs_bf  = (__bf16*)(bp);                             // 16MB
    __bf16* wbig   = (__bf16*)(bp + (size_t)SZ * 2);            // 8MB (reused)
    __bf16* wsmall = (__bf16*)(bp + (size_t)SZ * 2 + (size_t)SZW * 2); // 512KB
    __bf16* act_bf = (__bf16*)(bp + (size_t)SZ * 2 + (size_t)SZW * 2
                                  + (size_t)SZS * 2);           // 16MB

    const dim3 blk(256);
    const dim3 gP(M / 128, P_ / 64);                // N = 2048, NT=4
    const dim3 gD(M / 128, D_ / 64);                // N = 128,  NT=4
    const dim3 gH(M / 128, 1);                      // N = 16,   NT=1
    auto cvb = [](long n) { return dim3((unsigned)((n / 8 + 255) / 256)); };

    // 0) pack hs to bf16 once
    f32_to_bf16_kernel<<<cvb(SZ), blk, 0, stream>>>(hs, hs_bf, SZ);

    // 1) q/k/v projections (weight slot reused serially on the stream)
    f32_to_bf16_kernel<<<cvb(SZW), blk, 0, stream>>>(Wq, wbig, SZW);
    gemm_bf16_wmma<4><<<gP, blk, 0, stream>>>(hs_bf, wbig, bufA, M, P_, HID_);
    f32_to_bf16_kernel<<<cvb(SZW), blk, 0, stream>>>(Wk, wbig, SZW);
    gemm_bf16_wmma<4><<<gP, blk, 0, stream>>>(hs_bf, wbig, bufB, M, P_, HID_);
    f32_to_bf16_kernel<<<cvb(SZW), blk, 0, stream>>>(Wv, wbig, SZW);
    gemm_bf16_wmma<4><<<gP, blk, 0, stream>>>(hs_bf, wbig, bufC, M, P_, HID_);

    // 2) depthwise causal conv + silu
    const int nbConv = (int)((SZ + 255) / 256);
    conv_silu_kernel<<<nbConv, blk, 0, stream>>>(bufA, wq_conv, qc);
    conv_silu_kernel<<<nbConv, blk, 0, stream>>>(bufB, wk_conv, kc);
    conv_silu_kernel<<<nbConv, blk, 0, stream>>>(bufC, wv_conv, vc);

    // 3) decay gate: fa = hs@Wfa^T ; glin = fa@Wfb^T ; g = -exp(A)*softplus(+b)
    f32_to_bf16_kernel<<<cvb((long)D_ * HID_), blk, 0, stream>>>(Wfa, wsmall, (long)D_ * HID_);
    gemm_bf16_wmma<4><<<gD, blk, 0, stream>>>(hs_bf, wsmall, fbuf, M, D_, HID_);
    f32_to_bf16_kernel<<<cvb(SZF), blk, 0, stream>>>(fbuf, act_bf, SZF);
    f32_to_bf16_kernel<<<cvb(SZS), blk, 0, stream>>>(Wfb, wsmall, SZS);
    gemm_bf16_wmma<4><<<gP, blk, 0, stream>>>(act_bf, wsmall, bufC, M, P_, D_);
    kda_gate_kernel<<<nbConv, blk, 0, stream>>>(bufC, dt_bias, A_log);

    // 4) beta = sigmoid(hs @ Wb^T)
    f32_to_bf16_kernel<<<cvb((long)H_ * HID_), blk, 0, stream>>>(Wb, wsmall, (long)H_ * HID_);
    gemm_bf16_wmma<1><<<gH, blk, 0, stream>>>(hs_bf, wsmall, bbuf, M, H_, HID_);
    sigmoid_kernel<<<(M * H_ + 255) / 256, blk, 0, stream>>>(bbuf, M * H_);

    // 5) q/k L2 norm (q also scaled by D^-0.5)
    l2norm_kernel<<<M * H_, 128, 0, stream>>>(qc, 0.08838834764831845f);
    l2norm_kernel<<<M * H_, 128, 0, stream>>>(kc, 1.0f);

    // 6) output gate: ga = hs@Wga^T ; gout = ga@Wgb^T   (into bufA, now free)
    f32_to_bf16_kernel<<<cvb((long)D_ * HID_), blk, 0, stream>>>(Wga, wsmall, (long)D_ * HID_);
    gemm_bf16_wmma<4><<<gD, blk, 0, stream>>>(hs_bf, wsmall, fbuf, M, D_, HID_);
    f32_to_bf16_kernel<<<cvb(SZF), blk, 0, stream>>>(fbuf, act_bf, SZF);
    f32_to_bf16_kernel<<<cvb(SZS), blk, 0, stream>>>(Wgb, wsmall, SZS);
    gemm_bf16_wmma<4><<<gP, blk, 0, stream>>>(act_bf, wsmall, bufA, M, P_, D_);

    // 7) gated delta-rule recurrence (o into bufB, now free)
    recurrence_kernel<<<B_ * H_, 256, 0, stream>>>(qc, kc, vc, bufC, bbuf, bufB);

    // 8) gated RMS norm (in-place on bufB)
    normgate_kernel<<<M * H_, 128, 0, stream>>>(bufB, bufA, o_norm_w);

    // 9) output projection: out = o_n @ Wo^T
    f32_to_bf16_kernel<<<cvb(SZ), blk, 0, stream>>>(bufB, act_bf, SZ);
    f32_to_bf16_kernel<<<cvb(SZW), blk, 0, stream>>>(Wo, wbig, SZW);
    gemm_bf16_wmma<4><<<gP, blk, 0, stream>>>(act_bf, wbig, out, M, HID_, P_);
}
